// STMBlock_25726854103531
// MI455X (gfx1250) — matrix operand order repaired and checked by reading, not compiled, not run on hardware
//
#include <hip/hip_runtime.h>
#include <hip/hip_bf16.h>
#include <math.h>

// ---------------- problem constants ----------------
#define Bb 32
#define Nn 1024
#define Dd 768
#define Mm 256
#define Hh 12
#define HDd 64

typedef __attribute__((ext_vector_type(16))) __bf16 v16bf;
typedef __attribute__((ext_vector_type(8)))  __bf16 v8bf;
typedef __attribute__((ext_vector_type(8)))  float  v8f;

// ---------------- WMMA helpers (CDNA5, wave32) ----------------
__device__ inline v8f wmma_bf16(v16bf a, v16bf b, v8f c) {
  return __builtin_amdgcn_wmma_f32_16x16x32_bf16(false, a, false, b, (short)0, c, false, false);
}

// A-matrix 16x32 bf16 fragment, row-major source (lda in elements).
// Layout: lanes 0-15 hold K=0..15 for row m=lane;
// lanes 16-31 hold K=16..31 (matches load_b_frag).
__device__ inline v16bf load_a_frag(const __bf16* A, int lda) {
  int lane = threadIdx.x & 31;
  int m  = lane & 15;
  int kb = (lane >> 4) ? 16 : 0;
  const __bf16* p = A + (size_t)m * lda + kb;
  v8bf lo = *(const v8bf*)(p);
  v8bf hi = *(const v8bf*)(p + 8);
  v16bf r;
#pragma unroll
  for (int i = 0; i < 8; ++i) { r[i] = lo[i]; r[i + 8] = hi[i]; }
  return r;
}

// B-matrix 32x16 bf16 fragment from transposed (N x K) storage:
// lanes 0-15: column n=lane, K=0..15; lanes 16-31: n=lane-16, K=16..31.
__device__ inline v16bf load_b_frag(const __bf16* Bt, int ldb) {
  int lane = threadIdx.x & 31;
  int n  = lane & 15;
  int kb = (lane >> 4) ? 16 : 0;
  return *(const v16bf*)(Bt + (size_t)n * ldb + kb);
}

// ---------------- CDNA5 async Global->LDS copy (ASYNCcnt-tracked) ----------------
// LDS address: generic shared pointer truncated to 32 bits.
__device__ inline unsigned lds_off(const void* p) { return (unsigned)(uintptr_t)p; }

__device__ inline void async_copy_b128(unsigned lds_addr, const void* gaddr) {
  asm volatile("global_load_async_to_lds_b128 %0, %1, off"
               :: "v"(lds_addr), "v"(gaddr) : "memory");
}
__device__ inline void wait_async_le2() { asm volatile("s_wait_asynccnt 0x2" ::: "memory"); }
__device__ inline void wait_async_le0() { asm volatile("s_wait_asynccnt 0x0" ::: "memory"); }

// ---------------- small utility kernels ----------------
__global__ void zero_kernel(float* p, size_t n) {
  size_t i = (size_t)blockIdx.x * blockDim.x + threadIdx.x;
  for (; i < n; i += (size_t)gridDim.x * blockDim.x) p[i] = 0.0f;
}

__global__ void cvt_bf16_kernel(const float* __restrict__ a, __bf16* __restrict__ o, size_t n) {
  size_t i = (size_t)blockIdx.x * blockDim.x + threadIdx.x;
  if (i < n) o[i] = (__bf16)a[i];
}

// W[K][Nc] fp32  ->  Wt[Nc][K] bf16 (so B-fragments are contiguous in K)
__global__ void wconv_kernel(const float* __restrict__ W, __bf16* __restrict__ Wt, int Kd, int Nc) {
  int idx = blockIdx.x * blockDim.x + threadIdx.x;
  if (idx >= Kd * Nc) return;
  int k = idx / Nc, n = idx % Nc;
  Wt[(size_t)n * Kd + k] = (__bf16)W[idx];
}

__device__ inline float block_reduce_sum(float v, float* buf) {
  int t = threadIdx.x;
  buf[t] = v; __syncthreads();
  for (int s = blockDim.x >> 1; s > 0; s >>= 1) {
    if (t < s) buf[t] += buf[t + s];
    __syncthreads();
  }
  float r = buf[0]; __syncthreads();
  return r;
}

// ---------------- LayerNorm (+ optional score / x2 / tw / fp32 out) ----------------
__global__ void ln_kernel(const float* __restrict__ X, const float* __restrict__ g,
                          const float* __restrict__ beta, const float* __restrict__ sw,
                          const float* __restrict__ sb,
                          float* __restrict__ xn_f32, __bf16* __restrict__ xn_bf,
                          float* __restrict__ x2, float* __restrict__ score,
                          float* __restrict__ tw, int D_) {
  __shared__ float buf[256];
  int row = blockIdx.x;
  const float* xr = X + (size_t)row * D_;
  float s = 0.f, ss = 0.f;
  for (int i = threadIdx.x; i < D_; i += blockDim.x) { float v = xr[i]; s += v; ss += v * v; }
  s  = block_reduce_sum(s, buf);
  ss = block_reduce_sum(ss, buf);
  float mu = s / D_;
  float var = ss / D_ - mu * mu;
  float rstd = rsqrtf(var + 1e-5f);
  float dot = 0.f, sq = 0.f;
  for (int i = threadIdx.x; i < D_; i += blockDim.x) {
    float v = (xr[i] - mu) * rstd * g[i] + beta[i];
    if (xn_f32) xn_f32[(size_t)row * D_ + i] = v;
    xn_bf[(size_t)row * D_ + i] = (__bf16)v;
    if (sw) { dot += v * sw[i]; sq += v * v; }
  }
  if (sw) {
    dot = block_reduce_sum(dot, buf);
    sq  = block_reduce_sum(sq, buf);
    if (threadIdx.x == 0) {
      float sc = dot + sb[0];
      score[row] = sc;
      tw[row] = __expf(sc);
      x2[row] = sq;
    }
  }
}

// ---------------- Gram matrix -> pairwise distance (WMMA + async LDS staging) ----------------
// B-operand tile (rows of X for the tj block) is shared by all 4 waves: stage it in
// LDS with async copies, double-buffered over K-chunks of 128.
__global__ void __launch_bounds__(128) gram_dist_kernel(const __bf16* __restrict__ Xb,
                                                        const float* __restrict__ x2,
                                                        float* __restrict__ dist) {
  __shared__ __align__(32) __bf16 btile[2][16 * 128];
  int b  = blockIdx.z;
  int tj = blockIdx.x;
  int ti = blockIdx.y * blockDim.y + threadIdx.y;
  int tid = threadIdx.y * 32 + threadIdx.x;
  const __bf16* Ai = Xb + ((size_t)b * Nn + (size_t)ti * 16) * Dd;
  const __bf16* Aj = Xb + ((size_t)b * Nn + (size_t)tj * 16) * Dd;  // B = X^T
  unsigned lb = lds_off(&btile[0][0]);

  // stage(buf, k0): 16 rows x 128 halfs = 256 x 16B, 2 async ops per thread
#define GD_STAGE(bufi, k0)                                                         \
  {                                                                                \
    int c0 = tid, r0 = c0 >> 4, col0 = (c0 & 15) * 8;                              \
    async_copy_b128(lb + (unsigned)((((bufi) << 11) + r0 * 128 + col0) * 2),       \
                    Aj + (size_t)r0 * Dd + (k0) + col0);                           \
    int c1 = tid + 128, r1 = c1 >> 4, col1 = (c1 & 15) * 8;                        \
    async_copy_b128(lb + (unsigned)((((bufi) << 11) + r1 * 128 + col1) * 2),       \
                    Aj + (size_t)r1 * Dd + (k0) + col1);                           \
  }

  GD_STAGE(0, 0)
  v8f acc0 = {}, acc1 = {};
  const int nch = Dd >> 7;   // 6 chunks of 128
  for (int ch = 0; ch < nch; ++ch) {
    int k0 = ch << 7;
    if (ch + 1 < nch) {
      GD_STAGE((ch + 1) & 1, k0 + 128)
      __builtin_prefetch(Ai + (size_t)(threadIdx.x & 15) * Dd + k0 + 128, 0, 0);
      wait_async_le2();
    } else {
      wait_async_le0();
    }
    __syncthreads();
    const __bf16* bt = &btile[ch & 1][0];
    acc0 = wmma_bf16(load_a_frag(Ai + k0,      Dd), load_b_frag(bt,      128), acc0);
    acc1 = wmma_bf16(load_a_frag(Ai + k0 + 32, Dd), load_b_frag(bt + 32, 128), acc1);
    acc0 = wmma_bf16(load_a_frag(Ai + k0 + 64, Dd), load_b_frag(bt + 64, 128), acc0);
    acc1 = wmma_bf16(load_a_frag(Ai + k0 + 96, Dd), load_b_frag(bt + 96, 128), acc1);
    __syncthreads();
  }
#undef GD_STAGE

  int lane = threadIdx.x & 31;
  int n = lane & 15, mb = (lane >> 4) ? 8 : 0;
  int gj = tj * 16 + n;
  float x2j = x2[(size_t)b * Nn + gj];
  const float inv = rsqrtf((float)Dd);
  float* dr = dist + (size_t)b * Nn * Nn;
#pragma unroll
  for (int r = 0; r < 8; ++r) {
    int gi = ti * 16 + mb + r;
    float d2 = x2[(size_t)b * Nn + gi] + x2j - 2.0f * (acc0[r] + acc1[r]);
    dr[(size_t)gi * Nn + gj] = sqrtf(fmaxf(d2, 0.0f)) * inv;
  }
}

// ---------------- density: exp(-mean of 5 smallest dist^2), + per-batch max ----------------
__global__ void density_kernel(const float* __restrict__ dist, float* __restrict__ density,
                               int* __restrict__ distmax_i) {
  __shared__ float sval[256];
  __shared__ int   sidx[256];
  int row = blockIdx.x;               // b*N + i
  int b = row / Nn;
  int t = threadIdx.x;
  const float* dr = dist + (size_t)row * Nn;
  float dv[4];
  float lmax = 0.f;
#pragma unroll
  for (int u = 0; u < 4; ++u) { dv[u] = dr[t + u * 256]; lmax = fmaxf(lmax, dv[u]); }
  sval[t] = lmax; __syncthreads();
  for (int s = 128; s > 0; s >>= 1) { if (t < s) sval[t] = fmaxf(sval[t], sval[t + s]); __syncthreads(); }
  if (t == 0) atomicMax(&distmax_i[b], __float_as_int(sval[0]));  // dist >= 0
  __syncthreads();
  float acc = 0.f;
  for (int round = 0; round < 5; ++round) {
    float lmin = 3.4e38f; int lidx = 0;
#pragma unroll
    for (int u = 0; u < 4; ++u) if (dv[u] < lmin) { lmin = dv[u]; lidx = u; }
    sval[t] = lmin; sidx[t] = t * 4 + lidx; __syncthreads();
    for (int s = 128; s > 0; s >>= 1) {
      if (t < s && sval[t + s] < sval[t]) { sval[t] = sval[t + s]; sidx[t] = sidx[t + s]; }
      __syncthreads();
    }
    float mv = sval[0]; int widx = sidx[0];
    acc += mv * mv;
    if (t == (widx >> 2)) dv[widx & 3] = 3.4e38f;
    __syncthreads();
  }
  if (t == 0) density[row] = __expf(-acc * 0.2f);
}

// ---------------- dmin * density (cluster score) ----------------
__global__ void dmin_kernel(const float* __restrict__ dist, const float* __restrict__ density,
                            const int* __restrict__ distmax_i, float* __restrict__ cscore) {
  __shared__ float buf[256];
  int row = blockIdx.x; int b = row / Nn; int i = row % Nn;
  float di = density[row];
  const float* dr  = dist + (size_t)row * Nn;
  const float* den = density + (size_t)b * Nn;
  float lmin = __int_as_float(distmax_i[b]);
  for (int j = threadIdx.x; j < Nn; j += blockDim.x) {
    float dj = den[j];
    bool higher = (dj > di) || (dj == di && j < i);   // deterministic tiebreak
    if (higher) lmin = fminf(lmin, dr[j]);
  }
  buf[threadIdx.x] = lmin; __syncthreads();
  for (int s = 128; s > 0; s >>= 1) { if (threadIdx.x < s) buf[threadIdx.x] = fminf(buf[threadIdx.x], buf[threadIdx.x + s]); __syncthreads(); }
  if (threadIdx.x == 0) cscore[row] = buf[0] * di;
}

// ---------------- top-M via in-LDS bitonic sort (descending) ----------------
__global__ void topm_kernel(const float* __restrict__ cscore, int* __restrict__ index_down) {
  __shared__ float v[Nn];
  __shared__ int   ix[Nn];
  int b = blockIdx.x, t = threadIdx.x;    // 512 threads
  for (int u = 0; u < 2; ++u) { int i = t + u * 512; v[i] = cscore[(size_t)b * Nn + i]; ix[i] = i; }
  __syncthreads();
  for (int k = 2; k <= Nn; k <<= 1) {
    for (int j = k >> 1; j > 0; j >>= 1) {
      for (int u = 0; u < 2; ++u) {
        int i = t + u * 512;
        int p = i ^ j;
        if (p > i) {
          bool desc = ((i & k) == 0);
          bool sw = desc ? (v[i] < v[p]) : (v[i] > v[p]);
          if (sw) {
            float tv = v[i]; v[i] = v[p]; v[p] = tv;
            int  ti = ix[i]; ix[i] = ix[p]; ix[p] = ti;
          }
        }
      }
      __syncthreads();
    }
  }
  if (t < Mm) index_down[b * Mm + t] = ix[t];
}

// ---------------- assign each token to nearest selected center ----------------
__global__ void assign_kernel(const float* __restrict__ dist, const int* __restrict__ index_down,
                              int* __restrict__ idx_cluster) {
  int b = blockIdx.y;
  int n = blockIdx.x * blockDim.x + threadIdx.x;
  const float* db = dist + (size_t)b * Nn * Nn;
  const int* idb = index_down + b * Mm;
  float best = 3.4e38f; int bm = 0;
  for (int m = 0; m < Mm; ++m) {
    float d = db[(size_t)idb[m] * Nn + n];
    if (d < best) { best = d; bm = m; }
  }
  idx_cluster[(size_t)b * Nn + n] = bm;
}

__global__ void assign_override_kernel(const int* __restrict__ index_down, int* __restrict__ idx_cluster) {
  int b = blockIdx.x, m = threadIdx.x;
  idx_cluster[(size_t)b * Nn + index_down[b * Mm + m]] = m;
}

// ---------------- merger (weighted scatter-mean) ----------------
__global__ void segw_kernel(const int* __restrict__ idx_cluster, const float* __restrict__ tw,
                            float* __restrict__ all_w) {
  int t = blockIdx.x * blockDim.x + threadIdx.x;     // over B*N
  int b = t / Nn;
  atomicAdd(&all_w[b * Mm + idx_cluster[t]], tw[t]);
}

__global__ void merge_kernel(const float* __restrict__ xn, const int* __restrict__ idx_cluster,
                             const float* __restrict__ tw, const float* __restrict__ all_w,
                             float* __restrict__ merged) {
  int row = blockIdx.x; int b = row / Nn;
  int c = idx_cluster[row];
  float w = tw[row] / (all_w[b * Mm + c] + 1e-6f);
  const float* xr = xn + (size_t)row * Dd;
  float* mr = merged + ((size_t)b * Mm + c) * Dd;
  for (int i = threadIdx.x; i < Dd; i += blockDim.x) atomicAdd(&mr[i], xr[i] * w);
}

// ---------------- generic batched GEMM: C = epilogue(A[b] * Wt^T) ----------------
// A: [batch, R, Kd] bf16 row-major; Wt: [Nc, Kd] bf16 (shared across batch).
// Weight tile (16 x Kd slab) is shared by all 4 waves -> staged through LDS with
// CDNA5 async copies, double-buffered across K-chunks of 128.
__global__ void __launch_bounds__(128) gemm_bf16_kernel(
    const __bf16* __restrict__ A, const __bf16* __restrict__ Bt,
    const float* __restrict__ bias, const float* __restrict__ resid,
    float* __restrict__ Cf, __bf16* __restrict__ Cbf,
    __bf16* __restrict__ Cbt, int R, int Nc, int Kd, int gelu) {
  __shared__ __align__(32) __bf16 btile[2][16 * 128];
  int b  = blockIdx.z;
  int tn = blockIdx.x;
  int tm = blockIdx.y * blockDim.y + threadIdx.y;
  int tid = threadIdx.y * 32 + threadIdx.x;
  const __bf16* Ab  = A  + (size_t)b * R * Kd + (size_t)tm * 16 * Kd;
  const __bf16* Btn = Bt + (size_t)tn * 16 * Kd;
  unsigned lb = lds_off(&btile[0][0]);

#define GE_STAGE(bufi, k0)                                                         \
  {                                                                                \
    int c0 = tid, r0 = c0 >> 4, col0 = (c0 & 15) * 8;                              \
    async_copy_b128(lb + (unsigned)((((bufi) << 11) + r0 * 128 + col0) * 2),       \
                    Btn + (size_t)r0 * Kd + (k0) + col0);                          \
    int c1 = tid + 128, r1 = c1 >> 4, col1 = (c1 & 15) * 8;                        \
    async_copy_b128(lb + (unsigned)((((bufi) << 11) + r1 * 128 + col1) * 2),       \
                    Btn + (size_t)r1 * Kd + (k0) + col1);                          \
  }

  GE_STAGE(0, 0)
  v8f acc0 = {}, acc1 = {};
  const int nch = Kd >> 7;
  for (int ch = 0; ch < nch; ++ch) {
    int k0 = ch << 7;
    if (ch + 1 < nch) {
      GE_STAGE((ch + 1) & 1, k0 + 128)
      __builtin_prefetch(Ab + (size_t)(threadIdx.x & 15) * Kd + k0 + 128, 0, 0);
      wait_async_le2();
    } else {
      wait_async_le0();
    }
    __syncthreads();
    const __bf16* bt = &btile[ch & 1][0];
    acc0 = wmma_bf16(load_a_frag(Ab + k0,      Kd), load_b_frag(bt,      128), acc0);
    acc1 = wmma_bf16(load_a_frag(Ab + k0 + 32, Kd), load_b_frag(bt + 32, 128), acc1);
    acc0 = wmma_bf16(load_a_frag(Ab + k0 + 64, Kd), load_b_frag(bt + 64, 128), acc0);
    acc1 = wmma_bf16(load_a_frag(Ab + k0 + 96, Kd), load_b_frag(bt + 96, 128), acc1);
    __syncthreads();
  }
#undef GE_STAGE

  int lane = threadIdx.x & 31;
  int n = lane & 15, mb = (lane >> 4) ? 8 : 0;
  int gn = tn * 16 + n;
  float bv = bias ? bias[gn] : 0.0f;
#pragma unroll
  for (int r = 0; r < 8; ++r) {
    int gm = tm * 16 + mb + r;
    size_t o = (size_t)b * R * Nc + (size_t)gm * Nc + gn;
    float v = acc0[r] + acc1[r] + bv;
    if (resid) v += resid[o];
    if (gelu)  v = 0.5f * v * (1.0f + erff(v * 0.70710678118f));
    if (Cf)  Cf[o]  = v;
    if (Cbf) Cbf[o] = (__bf16)v;
    if (Cbt) Cbt[(size_t)b * R * Nc + (size_t)gn * R + gm] = (__bf16)v;  // [b, Nc, R]
  }
}

// ---------------- flash cross-attention (WMMA QK^T and PV, online softmax) ----------------
__global__ void flash_attn_kernel(const __bf16* __restrict__ Q, const __bf16* __restrict__ Kb,
                                  const __bf16* __restrict__ Vt, const float* __restrict__ bias,
                                  __bf16* __restrict__ O, float scale) {
  __shared__ __align__(32) __bf16 pshare[4][16 * 32];
  int b = blockIdx.z, h = blockIdx.y;
  int mt = blockIdx.x * blockDim.y + threadIdx.y;
  int lane = threadIdx.x & 31;
  __bf16* pl = pshare[threadIdx.y];

  const __bf16* Qb  = Q  + ((size_t)b * Mm + (size_t)mt * 16) * Dd + h * HDd;
  const __bf16* Kbb = Kb + (size_t)b * Nn * Dd + h * HDd;
  const __bf16* Vb  = Vt + (size_t)b * Dd * Nn + (size_t)(h * HDd) * Nn;
  const float*  bs  = bias + (size_t)b * Nn;

  v16bf qa0 = load_a_frag(Qb,      Dd);
  v16bf qa1 = load_a_frag(Qb + 32, Dd);

  v8f o0 = {}, o1 = {}, o2 = {}, o3 = {};
  float m8[8], l8[8], al[8];
#pragma unroll
  for (int r = 0; r < 8; ++r) { m8[r] = -3.0e38f; l8[r] = 0.0f; }

  int n = lane & 15, mb = (lane >> 4) ? 8 : 0;

  for (int j0 = 0; j0 < Nn; j0 += 32) {
    v8f s0 = {}, s1 = {};
    s0 = wmma_bf16(qa0, load_b_frag(Kbb + (size_t)j0 * Dd,        Dd), s0);
    s0 = wmma_bf16(qa1, load_b_frag(Kbb + (size_t)j0 * Dd + 32,   Dd), s0);
    s1 = wmma_bf16(qa0, load_b_frag(Kbb + (size_t)(j0 + 16) * Dd,      Dd), s1);
    s1 = wmma_bf16(qa1, load_b_frag(Kbb + (size_t)(j0 + 16) * Dd + 32, Dd), s1);
    float b0 = bs[j0 + n], b1 = bs[j0 + 16 + n];
#pragma unroll
    for (int r = 0; r < 8; ++r) {
      float v0 = s0[r] * scale + b0;
      float v1 = s1[r] * scale + b1;
      float mx = fmaxf(v0, v1);
      mx = fmaxf(mx, __shfl_xor(mx, 1, 32));
      mx = fmaxf(mx, __shfl_xor(mx, 2, 32));
      mx = fmaxf(mx, __shfl_xor(mx, 4, 32));
      mx = fmaxf(mx, __shfl_xor(mx, 8, 32));
      float mn = fmaxf(m8[r], mx);
      float a  = __expf(m8[r] - mn);
      v0 = __expf(v0 - mn);
      v1 = __expf(v1 - mn);
      float rs = v0 + v1;
      rs += __shfl_xor(rs, 1, 32);
      rs += __shfl_xor(rs, 2, 32);
      rs += __shfl_xor(rs, 4, 32);
      rs += __shfl_xor(rs, 8, 32);
      l8[r] = l8[r] * a + rs;
      m8[r] = mn;
      al[r] = a;
      // stage P (C-layout -> row-major LDS tile 16x32) for A-fragment reuse
      pl[(mb + r) * 32 + n]      = (__bf16)v0;
      pl[(mb + r) * 32 + 16 + n] = (__bf16)v1;
    }
#pragma unroll
    for (int r = 0; r < 8; ++r) { o0[r] *= al[r]; o1[r] *= al[r]; o2[r] *= al[r]; o3[r] *= al[r]; }
    __syncthreads();
    v16bf pa = load_a_frag(pl, 32);
    o0 = wmma_bf16(pa, load_b_frag(Vb + (size_t)0  * Nn + j0, Nn), o0);
    o1 = wmma_bf16(pa, load_b_frag(Vb + (size_t)16 * Nn + j0, Nn), o1);
    o2 = wmma_bf16(pa, load_b_frag(Vb + (size_t)32 * Nn + j0, Nn), o2);
    o3 = wmma_bf16(pa, load_b_frag(Vb + (size_t)48 * Nn + j0, Nn), o3);
    __syncthreads();
  }
#pragma unroll
  for (int r = 0; r < 8; ++r) {
    int gm = mt * 16 + mb + r;
    float invl = 1.0f / l8[r];
    size_t base = ((size_t)b * Mm + gm) * Dd + h * HDd + n;
    O[base + 0]  = (__bf16)(o0[r] * invl);
    O[base + 16] = (__bf16)(o1[r] * invl);
    O[base + 32] = (__bf16)(o2[r] * invl);
    O[base + 48] = (__bf16)(o3[r] * invl);
  }
}

// ============================================================================
extern "C" void kernel_launch(void* const* d_in, const int* in_sizes, int n_in,
                              void* d_out, int out_size, void* d_ws, size_t ws_size,
                              hipStream_t stream) {
  (void)in_sizes; (void)n_in; (void)out_size; (void)ws_size;
  const float* x       = (const float*)d_in[0];
  const float* norm1_g = (const float*)d_in[1];
  const float* norm1_b = (const float*)d_in[2];
  const float* score_w = (const float*)d_in[3];
  const float* score_b = (const float*)d_in[4];
  const float* wq      = (const float*)d_in[5];
  const float* wk      = (const float*)d_in[6];
  const float* wv      = (const float*)d_in[7];
  const float* proj_w  = (const float*)d_in[8];
  const float* proj_b  = (const float*)d_in[9];
  const float* norm2_g = (const float*)d_in[10];
  const float* norm2_b = (const float*)d_in[11];
  const float* fc1_w   = (const float*)d_in[12];
  const float* fc1_b   = (const float*)d_in[13];
  const float* fc2_w   = (const float*)d_in[14];
  const float* fc2_b   = (const float*)d_in[15];
  float* out = (float*)d_out;

  // ---- workspace bump allocator ----
  char* base = (char*)d_ws;
  size_t off = 0;
  auto alloc = [&](size_t bytes) -> void* {
    void* p = base + off;
    off = (off + bytes + 255) & ~(size_t)255;
    return p;
  };
  const size_t BN = (size_t)Bb * Nn, BND = BN * Dd;
  const size_t BM = (size_t)Bb * Mm, BMD = BM * Dd;

  float*  xn_f32   = (float*)alloc(BND * 4);
  __bf16* xn_bf    = (__bf16*)alloc(BND * 2);
  float*  x2buf    = (float*)alloc(BN * 4);
  float*  score    = (float*)alloc(BN * 4);
  float*  tw       = (float*)alloc(BN * 4);
  float*  dist     = (float*)alloc((size_t)Bb * Nn * Nn * 4);
  float*  density  = (float*)alloc(BN * 4);
  int*    distmaxi = (int*)alloc(Bb * 4);
  float*  cscore   = (float*)alloc(BN * 4);
  int*    idx_down = (int*)alloc(BM * 4);
  int*    idx_clu  = (int*)alloc(BN * 4);
  float*  all_w    = (float*)alloc(BM * 4);
  float*  merged   = (float*)alloc(BMD * 4);
  __bf16* merged_b = (__bf16*)alloc(BMD * 2);
  __bf16* qb       = (__bf16*)alloc(BMD * 2);
  __bf16* kb       = (__bf16*)alloc(BND * 2);
  __bf16* vt       = (__bf16*)alloc(BND * 2);
  __bf16* attn_o   = (__bf16*)alloc(BMD * 2);
  float*  feat     = (float*)alloc(BMD * 4);
  __bf16* h2_bf    = (__bf16*)alloc(BMD * 2);
  __bf16* g1_bf    = (__bf16*)alloc(BM * 4 * Dd * 2);
  __bf16* wq_t     = (__bf16*)alloc((size_t)Dd * Dd * 2);
  __bf16* wk_t     = (__bf16*)alloc((size_t)Dd * Dd * 2);
  __bf16* wv_t     = (__bf16*)alloc((size_t)Dd * Dd * 2);
  __bf16* proj_t   = (__bf16*)alloc((size_t)Dd * Dd * 2);
  __bf16* fc1_t    = (__bf16*)alloc((size_t)Dd * 4 * Dd * 2);
  __bf16* fc2_t    = (__bf16*)alloc((size_t)Dd * 4 * Dd * 2);

  // ---- zero accumulators ----
  zero_kernel<<<256, 256, 0, stream>>>(all_w, BM);
  zero_kernel<<<1024, 256, 0, stream>>>(merged, BMD);
  zero_kernel<<<1, 32, 0, stream>>>((float*)distmaxi, Bb);

  // ---- weight convert + transpose ----
  int wd = Dd * Dd, wf = Dd * 4 * Dd;
  wconv_kernel<<<(wd + 255) / 256, 256, 0, stream>>>(wq,     wq_t,   Dd,     Dd);
  wconv_kernel<<<(wd + 255) / 256, 256, 0, stream>>>(wk,     wk_t,   Dd,     Dd);
  wconv_kernel<<<(wd + 255) / 256, 256, 0, stream>>>(wv,     wv_t,   Dd,     Dd);
  wconv_kernel<<<(wd + 255) / 256, 256, 0, stream>>>(proj_w, proj_t, Dd,     Dd);
  wconv_kernel<<<(wf + 255) / 256, 256, 0, stream>>>(fc1_w,  fc1_t,  Dd,     4 * Dd);
  wconv_kernel<<<(wf + 255) / 256, 256, 0, stream>>>(fc2_w,  fc2_t,  4 * Dd, Dd);

  // ---- LN1 + token score ----
  ln_kernel<<<(int)BN, 256, 0, stream>>>(x, norm1_g, norm1_b, score_w, score_b,
                                         xn_f32, xn_bf, x2buf, score, tw, Dd);

  // ---- pairwise distances via WMMA Gram (async-LDS staged B tile) ----
  gram_dist_kernel<<<dim3(Nn / 16, Nn / 64, Bb), dim3(32, 4), 0, stream>>>(xn_bf, x2buf, dist);

  // ---- clustering ----
  density_kernel<<<(int)BN, 256, 0, stream>>>(dist, density, distmaxi);
  dmin_kernel<<<(int)BN, 256, 0, stream>>>(dist, density, distmaxi, cscore);
  topm_kernel<<<Bb, 512, 0, stream>>>(cscore, idx_down);
  assign_kernel<<<dim3(Nn / 256, Bb), 256, 0, stream>>>(dist, idx_down, idx_clu);
  assign_override_kernel<<<Bb, Mm, 0, stream>>>(idx_down, idx_clu);

  // ---- merger ----
  segw_kernel<<<(int)(BN / 256), 256, 0, stream>>>(idx_clu, tw, all_w);
  merge_kernel<<<(int)BN, 256, 0, stream>>>(xn_f32, idx_clu, tw, all_w, merged);
  cvt_bf16_kernel<<<(int)((BMD + 255) / 256), 256, 0, stream>>>(merged, merged_b, BMD);

  // ---- QKV projections (WMMA, async-LDS staged weights) ----
  gemm_bf16_kernel<<<dim3(Dd / 16, Mm / 64, Bb), dim3(32, 4), 0, stream>>>(
      merged_b, wq_t, nullptr, nullptr, nullptr, qb, nullptr, Mm, Dd, Dd, 0);
  gemm_bf16_kernel<<<dim3(Dd / 16, Nn / 64, Bb), dim3(32, 4), 0, stream>>>(
      xn_bf, wk_t, nullptr, nullptr, nullptr, kb, nullptr, Nn, Dd, Dd, 0);
  gemm_bf16_kernel<<<dim3(Dd / 16, Nn / 64, Bb), dim3(32, 4), 0, stream>>>(
      xn_bf, wv_t, nullptr, nullptr, nullptr, nullptr, vt, Nn, Dd, Dd, 0);  // transposed store

  // ---- biased flash attention (WMMA) ----
  float scale = rsqrtf((float)HDd);
  flash_attn_kernel<<<dim3(Mm / 64, Hh, Bb), dim3(32, 4), 0, stream>>>(
      qb, kb, vt, score, attn_o, scale);

  // ---- output projection + residual -> feat ----
  gemm_bf16_kernel<<<dim3(Dd / 16, Mm / 64, Bb), dim3(32, 4), 0, stream>>>(
      attn_o, proj_t, proj_b, merged, feat, nullptr, nullptr, Mm, Dd, Dd, 0);

  // ---- MLP ----
  ln_kernel<<<(int)BM, 256, 0, stream>>>(feat, norm2_g, norm2_b, nullptr, nullptr,
                                         nullptr, h2_bf, nullptr, nullptr, nullptr, Dd);
  gemm_bf16_kernel<<<dim3(4 * Dd / 16, Mm / 64, Bb), dim3(32, 4), 0, stream>>>(
      h2_bf, fc1_t, fc1_b, nullptr, nullptr, g1_bf, nullptr, Mm, 4 * Dd, Dd, 1);  // gelu
  gemm_bf16_kernel<<<dim3(Dd / 16, Mm / 64, Bb), dim3(32, 4), 0, stream>>>(
      g1_bf, fc2_t, fc2_b, feat, out, nullptr, nullptr, Mm, Dd, 4 * Dd, 0);
}
